// RobertaModel_84971632984569
// MI455X (gfx1250) — matrix operand order repaired and checked by reading, not compiled, not run on hardware
//
#include <hip/hip_runtime.h>

// ---------------------------------------------------------------------------
// RoBERTa-base forward for MI455X (gfx1250), wave32 + WMMA bf16 + TDM.
// Weight-bandwidth-bound at 23.3 TB/s -> bf16-resident weights/activations,
// GEMM tiles streamed by TENSOR_LOAD_TO_LDS with double buffering (TENSORcnt
// overlaps DMA with v_wmma_f32_16x16x32_bf16 math).
// ---------------------------------------------------------------------------

typedef __attribute__((ext_vector_type(16))) __bf16       v16bf;
typedef __attribute__((ext_vector_type(8)))  float        v8f;
typedef __attribute__((ext_vector_type(4)))  unsigned int v4u;
typedef __attribute__((ext_vector_type(8)))  int          v8i;
typedef __attribute__((ext_vector_type(4)))  int          v4i;

#define DEV __device__ __forceinline__

#if __has_builtin(__builtin_amdgcn_tensor_load_to_lds)
#define HAVE_TDM 1
#else
#define HAVE_TDM 0
#endif

DEV unsigned int lds_addr_of(const void* p) {
    // flat shared address: low 32 bits are the LDS byte offset
    return (unsigned int)(size_t)p;
}

// ---- Tensor Data Mover: 2D bf16 tile load, descriptor per ISA 8.3/8.4 -----
// Loads tile1 rows x tile0 contiguous elements from gptr (row stride stride0
// elements) into LDS at lds_addr, row-major [tile1][tile0].
#if HAVE_TDM
DEV void tdm_load_2d(unsigned int lds_addr, const void* gptr,
                     unsigned int tile0, unsigned int tile1,
                     unsigned int stride0) {
    unsigned long long ga = (unsigned long long)(size_t)gptr;
    v4u g0;
    g0.x = 1u;                                             // count=1 (valid)
    g0.y = lds_addr;                                       // LDS byte address
    g0.z = (unsigned int)ga;                               // global_addr lo
    g0.w = (unsigned int)((ga >> 32) & 0x01FFFFFFu) | (2u << 30); // hi | type=2
    v8i g1;
    g1[0] = (int)(1u << 16);                 // wg_mask=0, data_size=1 (2B)
    g1[1] = (int)((tile0 & 0xFFFFu) << 16);  // tensor_dim0 = tile0
    g1[2] = (int)((tile1 & 0xFFFFu) << 16);  // tensor_dim1 = tile1
    g1[3] = (int)(tile0 << 16);              // tile_dim0
    g1[4] = (int)tile1;                      // tile_dim1 (tile_dim2 = 0)
    g1[5] = (int)stride0;                    // tensor_dim0_stride lo
    g1[6] = 0;
    g1[7] = 0;
    v4i z4 = {0, 0, 0, 0};
#if __clang_major__ >= 23
    v8i z8 = {0, 0, 0, 0, 0, 0, 0, 0};
    __builtin_amdgcn_tensor_load_to_lds(g0, g1, z4, z4, z8, 0);
#else
    __builtin_amdgcn_tensor_load_to_lds(g0, g1, z4, z4, 0);
#endif
}
#endif

// ---- WMMA fragment loaders (contiguous k order shared by A and B, wave32) -------------------
// A 16x32 bf16: lane = row (lane&15); element i -> K = (lane>>4)*16 + i
// (sequential); pairs are K-contiguous -> 8 dword LDS loads.
DEV v16bf load_a_frag(const __bf16* base, int stride, int lane) {
    const int r = lane & 15;
    const int half = lane >> 4;
    const __bf16* row = base + r * stride;
    union { v16bf v; unsigned int u[8]; } f;
#pragma unroll
    for (int j = 0; j < 8; ++j) {
        int k = (half << 4) + (j << 1);
        f.u[j] = *(const unsigned int*)(row + k);
    }
    return f.v;
}

// B 32x16 bf16: lane = column (lane&15); element i -> K = (lane>>4)*16 + i
// (sequential).  base stored transposed [n][k] -> 8 contiguous dword loads.
DEV v16bf load_b_frag(const __bf16* base, int stride, int lane) {
    const int n = lane & 15;
    const int half = lane >> 4;
    const __bf16* col = base + n * stride + half * 16;
    union { v16bf v; unsigned int u[8]; } f;
#pragma unroll
    for (int j = 0; j < 8; ++j)
        f.u[j] = *(const unsigned int*)(col + 2 * j);
    return f.v;
}

DEV v8f wmma_bf16(v16bf a, v16bf b, v8f c) {
    return __builtin_amdgcn_wmma_f32_16x16x32_bf16(
        false, a, false, b, (short)0, c, false, false);
}

// ---- block reduction helper (256 threads) ---------------------------------
DEV float block_sum(float v, float* red) {
    int t = threadIdx.x;
    red[t] = v;
    __syncthreads();
#pragma unroll
    for (int s = 128; s > 0; s >>= 1) {
        if (t < s) red[t] += red[t + s];
        __syncthreads();
    }
    float r = red[0];
    __syncthreads();
    return r;
}

// ---------------------------------------------------------------------------
// Weight convert + transpose: W[K,N] fp32 -> WT[N,K] bf16 (32x32 LDS tiles).
// ---------------------------------------------------------------------------
__global__ __launch_bounds__(256) void transpose_bf16_kernel(
    const float* __restrict__ W, __bf16* __restrict__ WT, int K, int N)
{
    __shared__ __bf16 tile[32][33];
    const int n0 = blockIdx.x * 32;
    const int k0 = blockIdx.y * 32;
    const int c = threadIdx.x & 31;
    const int r0 = threadIdx.x >> 5;
#pragma unroll
    for (int p = 0; p < 4; ++p) {
        int k = r0 + p * 8;
        tile[k][c] = (__bf16)W[(size_t)(k0 + k) * N + n0 + c];
    }
    __syncthreads();
#pragma unroll
    for (int p = 0; p < 4; ++p) {
        int n = r0 + p * 8;
        WT[(size_t)(n0 + n) * K + k0 + c] = tile[c][n];
    }
}

// ---------------------------------------------------------------------------
// Embedding gather + fused LayerNorm; writes fp32 hidden + bf16 mirror.
// ---------------------------------------------------------------------------
__global__ __launch_bounds__(256) void embed_ln_kernel(
    const int* __restrict__ ids, const float* __restrict__ we,
    const float* __restrict__ pe, const float* __restrict__ te,
    const float* __restrict__ g, const float* __restrict__ bta,
    float* __restrict__ outf, __bf16* __restrict__ outb)
{
    constexpr int H = 768, S = 512;
    const int tok = blockIdx.x;
    const int spos = tok % S;
    const int id = ids[tok];
    const int t = threadIdx.x;
    __shared__ float red[256];

    const float* w = we + (size_t)id * H;
    const float* p = pe + (size_t)(spos + 2) * H;
    float e0 = w[t]       + p[t]       + te[t];
    float e1 = w[t + 256] + p[t + 256] + te[t + 256];
    float e2 = w[t + 512] + p[t + 512] + te[t + 512];

    float mu = block_sum(e0 + e1 + e2, red) * (1.f / 768.f);
    float d0 = e0 - mu, d1 = e1 - mu, d2 = e2 - mu;
    float var = block_sum(d0 * d0 + d1 * d1 + d2 * d2, red) * (1.f / 768.f);
    float inv = rsqrtf(var + 1e-5f);

    float r0 = d0 * inv * g[t]       + bta[t];
    float r1 = d1 * inv * g[t + 256] + bta[t + 256];
    float r2 = d2 * inv * g[t + 512] + bta[t + 512];
    float* o = outf + (size_t)tok * H;
    __bf16* ob = outb + (size_t)tok * H;
    o[t] = r0;          o[t + 256] = r1;            o[t + 512] = r2;
    ob[t] = (__bf16)r0; ob[t + 256] = (__bf16)r1;   ob[t + 512] = (__bf16)r2;
}

// ---------------------------------------------------------------------------
// LayerNorm (residual already folded in); fp32 + bf16 outputs.
// ---------------------------------------------------------------------------
__global__ __launch_bounds__(256) void ln_kernel(
    const float* __restrict__ in, const float* __restrict__ g,
    const float* __restrict__ bta, float* __restrict__ outf,
    __bf16* __restrict__ outb)
{
    constexpr int H = 768;
    const int tok = blockIdx.x;
    const int t = threadIdx.x;
    __shared__ float red[256];
    const float* x = in + (size_t)tok * H;

    float x0 = x[t], x1 = x[t + 256], x2 = x[t + 512];
    float mu = block_sum(x0 + x1 + x2, red) * (1.f / 768.f);
    float d0 = x0 - mu, d1 = x1 - mu, d2 = x2 - mu;
    float var = block_sum(d0 * d0 + d1 * d1 + d2 * d2, red) * (1.f / 768.f);
    float inv = rsqrtf(var + 1e-5f);

    float r0 = d0 * inv * g[t]       + bta[t];
    float r1 = d1 * inv * g[t + 256] + bta[t + 256];
    float r2 = d2 * inv * g[t + 512] + bta[t + 512];
    float* o = outf + (size_t)tok * H;
    __bf16* ob = outb + (size_t)tok * H;
    o[t] = r0;          o[t + 256] = r1;            o[t + 512] = r2;
    ob[t] = (__bf16)r0; ob[t + 256] = (__bf16)r1;   ob[t + 512] = (__bf16)r2;
}

// ---------------------------------------------------------------------------
// GEMM: C[M,N] = A[M,K] @ W[K,N] + bias, with W given transposed bf16 [N,K].
//   mode 0: bias -> bf16 out     mode 1: bias+residual -> fp32 out
//   mode 2: bias+GELU -> bf16 out
// Block tile 128x64, BK=32; double-buffered TDM streaming: iteration i waits
// on buffer cur, immediately issues the DMA for cur^1, then runs WMMA.
// ---------------------------------------------------------------------------
__global__ __launch_bounds__(256) void gemm_bias_kernel(
    const __bf16* __restrict__ A,   // [M,K] bf16
    const __bf16* __restrict__ WT,  // [N,K] bf16 (transposed weights)
    const float* __restrict__ bias,
    const float* __restrict__ residual,  // [M,N] fp32, mode 1 only
    void* __restrict__ out, int M, int N, int K, int mode)
{
    constexpr int BM = 128, BN = 64, BK = 32;
    __shared__ __bf16 As[2][BM * BK];   // [row][k]
    __shared__ __bf16 Bs[2][BN * BK];   // [n][k]

    const int tid = threadIdx.x;
    const int lane = tid & 31;
    const int wave = tid >> 5;
    const int m0 = blockIdx.y * BM;
    const int n0 = blockIdx.x * BN;

    v8f acc[4] = {};

#if HAVE_TDM
    if (tid == 0) {   // single issuer; TDM itself ignores EXEC
        tdm_load_2d(lds_addr_of(As[0]), A + (size_t)m0 * K, BK, BM, (unsigned)K);
        tdm_load_2d(lds_addr_of(Bs[0]), WT + (size_t)n0 * K, BK, BN, (unsigned)K);
    }
    int cur = 0;
    for (int k0 = 0; k0 < K; k0 += BK) {
        if (tid == 0)
            __builtin_amdgcn_s_wait_tensorcnt((unsigned short)0);
        __syncthreads();                        // buffer `cur` ready for all
        if (tid == 0 && k0 + BK < K) {          // stream next tile during math
            tdm_load_2d(lds_addr_of(As[cur ^ 1]),
                        A + (size_t)m0 * K + k0 + BK, BK, BM, (unsigned)K);
            tdm_load_2d(lds_addr_of(Bs[cur ^ 1]),
                        WT + (size_t)n0 * K + k0 + BK, BK, BN, (unsigned)K);
        }
        if (k0 + BK < K)   // keep next A tile warm near the WGP
            __builtin_prefetch(A + (size_t)(m0 + (tid & 127)) * K + k0 + BK, 0, 3);

        v16bf af = load_a_frag(As[cur] + (wave * 16) * BK, BK, lane);
#pragma unroll
        for (int nt = 0; nt < 4; ++nt) {
            v16bf bf = load_b_frag(Bs[cur] + (nt * 16) * BK, BK, lane);
            acc[nt] = wmma_bf16(af, bf, acc[nt]);
        }
        __syncthreads();                        // done with `cur` before reuse
        cur ^= 1;
    }
#else
    for (int k0 = 0; k0 < K; k0 += BK) {
        unsigned int* As32 = (unsigned int*)As[0];
        unsigned int* Bs32 = (unsigned int*)Bs[0];
        for (int idx = tid; idx < BM * (BK / 2); idx += 256) {
            int r = idx >> 4, c = idx & 15;
            As32[r * 16 + c] =
                ((const unsigned int*)(A + (size_t)(m0 + r) * K + k0))[c];
        }
        for (int idx = tid; idx < BN * (BK / 2); idx += 256) {
            int r = idx >> 4, c = idx & 15;
            Bs32[r * 16 + c] =
                ((const unsigned int*)(WT + (size_t)(n0 + r) * K + k0))[c];
        }
        __syncthreads();
        v16bf af = load_a_frag(As[0] + (wave * 16) * BK, BK, lane);
#pragma unroll
        for (int nt = 0; nt < 4; ++nt) {
            v16bf bf = load_b_frag(Bs[0] + (nt * 16) * BK, BK, lane);
            acc[nt] = wmma_bf16(af, bf, acc[nt]);
        }
        __syncthreads();
    }
#endif

    // Epilogue.  C/D layout: col = lane&15, row = (lane>>4)*8 + j.
    const int col16 = lane & 15;
    const int rbase = m0 + wave * 16 + ((lane >> 4) << 3);
#pragma unroll
    for (int nt = 0; nt < 4; ++nt) {
        int col = n0 + nt * 16 + col16;
        float bcol = bias[col];
#pragma unroll
        for (int j = 0; j < 8; ++j) {
            int row = rbase + j;
            float v = acc[nt][j] + bcol;
            size_t idx = (size_t)row * N + col;
            if (mode == 1) {
                ((float*)out)[idx] = v + residual[idx];
            } else if (mode == 2) {
                v = 0.5f * v * (1.0f + erff(v * 0.70710678118f));
                ((__bf16*)out)[idx] = (__bf16)v;
            } else {
                ((__bf16*)out)[idx] = (__bf16)v;
            }
        }
    }
}

// ---------------------------------------------------------------------------
// Attention: block per (b, head, 16-query tile).  bf16 q/k/v.  Q tile staged
// by TDM; QK^T and P@V on WMMA; softmax with wave32 shuffles.
// ---------------------------------------------------------------------------
__global__ __launch_bounds__(256) void attention_kernel(
    const __bf16* __restrict__ qp, const __bf16* __restrict__ kp,
    const __bf16* __restrict__ vp, __bf16* __restrict__ ctx)
{
    constexpr int S = 512, H = 768, DH = 64;
    const int qt = blockIdx.x;
    const int head = blockIdx.y;
    const int b = blockIdx.z;
    const int tid = threadIdx.x;
    const int lane = tid & 31;
    const int wave = tid >> 5;
    const int half = lane >> 4;
    const int l15 = lane & 15;
    const int q0 = qt * 16;
    const int hd = head * DH;
    const size_t bsH = (size_t)b * S * H;

    __shared__ __bf16 Qs[16 * 64];
    __shared__ float  Sc[16 * 512];
    __shared__ __bf16 Pb[16 * 512];

    // stage Q tile (16 rows x 64 dims, row stride H) via TDM
#if HAVE_TDM
    if (tid == 0) {
        tdm_load_2d(lds_addr_of(Qs), qp + bsH + (size_t)q0 * H + hd,
                    DH, 16, (unsigned)H);
        __builtin_amdgcn_s_wait_tensorcnt((unsigned short)0);
    }
#else
    for (int idx = tid; idx < 16 * 64; idx += 256) {
        int r = idx >> 6, d = idx & 63;
        Qs[r * 64 + d] = qp[bsH + (size_t)(q0 + r) * H + hd + d];
    }
#endif
    __syncthreads();

    // scores = (Q K^T) / 8;  wave w -> key tiles w, w+8, w+16, w+24
    for (int kt = wave; kt < 32; kt += 8) {
        v8f acc = {};
#pragma unroll
        for (int step = 0; step < 2; ++step) {
            v16bf af = load_a_frag(Qs + step * 32, 64, lane);
            // B = K^T: column = key row; K-index sequential -> 8 dword loads
            union { v16bf v; unsigned int u[8]; } bfr;
            const unsigned int* s32 = (const unsigned int*)
                (kp + bsH + (size_t)(kt * 16 + l15) * H + hd + step * 32 + half * 16);
#pragma unroll
            for (int j = 0; j < 8; ++j) bfr.u[j] = s32[j];
            acc = wmma_bf16(af, bfr.v, acc);
        }
        const int rbase = half << 3;
#pragma unroll
        for (int j = 0; j < 8; ++j)
            Sc[(rbase + j) * 512 + kt * 16 + l15] = acc[j] * 0.125f;
    }
    __syncthreads();

    // softmax per row (wave handles 2 rows)
#pragma unroll
    for (int rr = 0; rr < 2; ++rr) {
        const int row = wave * 2 + rr;
        float m = -3.4e38f;
#pragma unroll
        for (int t = 0; t < 16; ++t)
            m = fmaxf(m, Sc[row * 512 + lane + t * 32]);
#pragma unroll
        for (int off = 16; off > 0; off >>= 1)
            m = fmaxf(m, __shfl_xor(m, off, 32));
        float s = 0.f;
#pragma unroll
        for (int t = 0; t < 16; ++t) {
            float e = __expf(Sc[row * 512 + lane + t * 32] - m);
            Sc[row * 512 + lane + t * 32] = e;
            s += e;
        }
#pragma unroll
        for (int off = 16; off > 0; off >>= 1)
            s += __shfl_xor(s, off, 32);
        const float inv = 1.f / s;
#pragma unroll
        for (int t = 0; t < 16; ++t)
            Pb[row * 512 + lane + t * 32] =
                (__bf16)(Sc[row * 512 + lane + t * 32] * inv);
    }
    __syncthreads();

    // ctx(16x64) = P(16x512) @ V(512x64); waves 0..3 own 16-col tiles
    if (wave < 4) {
        const int dcol = hd + wave * 16 + l15;
        v8f acc = {};
        for (int ks = 0; ks < 16; ++ks) {
            v16bf af = load_a_frag(Pb + ks * 32, 512, lane);
            union { v16bf v; __bf16 e[16]; } bfr;
#pragma unroll
            for (int i = 0; i < 16; ++i) {
                int key = ks * 32 + half * 16 + i;
                bfr.e[i] = vp[bsH + (size_t)key * H + dcol];
            }
            acc = wmma_bf16(af, bfr.v, acc);
        }
        const int rbase = q0 + (half << 3);
#pragma unroll
        for (int j = 0; j < 8; ++j)
            ctx[bsH + (size_t)(rbase + j) * H + dcol] = (__bf16)acc[j];
    }
}

// ---------------------------------------------------------------------------
extern "C" void kernel_launch(void* const* d_in, const int* in_sizes, int n_in,
                              void* d_out, int out_size, void* d_ws, size_t ws_size,
                              hipStream_t stream) {
    constexpr int B = 16, S = 512, H = 768, L = 12, I = 3072;
    const int M = B * S;
    const size_t BSH = (size_t)M * H;
    const size_t BSI = (size_t)M * I;
    const size_t HH = (size_t)H * H;
    const size_t HI = (size_t)H * I;

    const int*   ids  = (const int*)  d_in[0];
    const float* we   = (const float*)d_in[1];
    const float* pe   = (const float*)d_in[2];
    const float* te   = (const float*)d_in[3];
    const float* lneg = (const float*)d_in[4];
    const float* lneb = (const float*)d_in[5];
    const float* Wq   = (const float*)d_in[6];
    const float* bq   = (const float*)d_in[7];
    const float* Wk   = (const float*)d_in[8];
    const float* bk   = (const float*)d_in[9];
    const float* Wv   = (const float*)d_in[10];
    const float* bv   = (const float*)d_in[11];
    const float* Wo   = (const float*)d_in[12];
    const float* bo   = (const float*)d_in[13];
    const float* g1   = (const float*)d_in[14];
    const float* b1   = (const float*)d_in[15];
    const float* Wi   = (const float*)d_in[16];
    const float* bi   = (const float*)d_in[17];
    const float* Wo2  = (const float*)d_in[18];
    const float* bo2  = (const float*)d_in[19];
    const float* g2   = (const float*)d_in[20];
    const float* b2   = (const float*)d_in[21];

    // workspace carve (~206 MiB)
    float*  x    = (float*)d_ws;        // hidden fp32 (residual source)
    float*  tmp  = x + BSH;             // pre-LN fp32
    float*  ao   = tmp + BSH;           // attn_out fp32
    __bf16* xb   = (__bf16*)(ao + BSH); // hidden bf16 (GEMM A input)
    __bf16* aob  = xb + BSH;            // attn_out bf16
    __bf16* qb   = aob + BSH;
    __bf16* kb   = qb + BSH;
    __bf16* vb   = kb + BSH;
    __bf16* cxb  = vb + BSH;            // attention context bf16
    __bf16* hb   = cxb + BSH;           // FFN intermediate bf16 [M,I]
    __bf16* wT   = hb + BSI;            // per-layer transposed bf16 weights

    const dim3 blk(256);
    const dim3 gHH(H / 64, M / 128);
    const dim3 gHI(I / 64, M / 128);
    const dim3 gAttn(S / 16, 12, B);

    embed_ln_kernel<<<M, blk, 0, stream>>>(ids, we, pe, te, lneg, lneb, x, xb);

    for (int l = 0; l < L; ++l) {
        const float* wq = Wq + (size_t)l * HH;
        const float* wk = Wk + (size_t)l * HH;
        const float* wv = Wv + (size_t)l * HH;
        const float* wo = Wo + (size_t)l * HH;
        const float* wi = Wi + (size_t)l * HI;
        const float* w2 = Wo2 + (size_t)l * HI;

        // convert+transpose this layer's weights to bf16 [N][K]
        transpose_bf16_kernel<<<dim3(24, 24), blk, 0, stream>>>(wq, wT + 0 * HH, H, H);
        transpose_bf16_kernel<<<dim3(24, 24), blk, 0, stream>>>(wk, wT + 1 * HH, H, H);
        transpose_bf16_kernel<<<dim3(24, 24), blk, 0, stream>>>(wv, wT + 2 * HH, H, H);
        transpose_bf16_kernel<<<dim3(24, 24), blk, 0, stream>>>(wo, wT + 3 * HH, H, H);
        transpose_bf16_kernel<<<dim3(96, 24), blk, 0, stream>>>(wi, wT + 4 * HH, H, I);
        transpose_bf16_kernel<<<dim3(24, 96), blk, 0, stream>>>(w2, wT + 4 * HH + HI, I, H);

        // QKV projections (bf16 out)
        gemm_bias_kernel<<<gHH, blk, 0, stream>>>(xb, wT + 0 * HH, bq + l * H, nullptr, qb, M, H, H, 0);
        gemm_bias_kernel<<<gHH, blk, 0, stream>>>(xb, wT + 1 * HH, bk + l * H, nullptr, kb, M, H, H, 0);
        gemm_bias_kernel<<<gHH, blk, 0, stream>>>(xb, wT + 2 * HH, bv + l * H, nullptr, vb, M, H, H, 0);

        attention_kernel<<<gAttn, blk, 0, stream>>>(qb, kb, vb, cxb);

        // output projection + residual(x) -> tmp fp32; LN1 -> ao/aob
        gemm_bias_kernel<<<gHH, blk, 0, stream>>>(cxb, wT + 3 * HH, bo + l * H, x, tmp, M, H, H, 1);
        ln_kernel<<<M, blk, 0, stream>>>(tmp, g1 + (size_t)l * H, b1 + (size_t)l * H, ao, aob);

        // FFN
        gemm_bias_kernel<<<gHI, blk, 0, stream>>>(aob, wT + 4 * HH, bi + (size_t)l * I, nullptr, hb, M, I, H, 2);
        gemm_bias_kernel<<<gHH, blk, 0, stream>>>(hb, wT + 4 * HH + HI, bo2 + l * H, ao, tmp, M, H, I, 1);

        // LN2 -> next hidden (fp32 + bf16); last layer fp32 goes to d_out
        float* xo = (l == L - 1) ? (float*)d_out : x;
        ln_kernel<<<M, blk, 0, stream>>>(tmp, g2 + (size_t)l * H, b2 + (size_t)l * H, xo, xb);
    }
}
